// Attention_16612933500950
// MI455X (gfx1250) — matrix operand order repaired and checked by reading, not compiled, not run on hardware
//
#include <hip/hip_runtime.h>
#include <hip/hip_bf16.h>
#include <stdint.h>

// ---------------------------------------------------------------------------
// Types for CDNA5 WMMA (wave32)
// ---------------------------------------------------------------------------
typedef __attribute__((ext_vector_type(16))) __bf16 v16bf;
typedef __attribute__((ext_vector_type(8)))  __bf16 v8bf;
typedef __attribute__((ext_vector_type(8)))  float  v8f;

#define BATCH 4
#define SEQ   2048
#define DIM   1024
#define HEADS 16
#define DHEAD 64

__device__ __forceinline__ unsigned short f2bfu(float f) {
    unsigned int u = __builtin_bit_cast(unsigned int, f);
    return (unsigned short)((u + 0x7FFFu + ((u >> 16) & 1u)) >> 16);
}

__device__ __forceinline__ v16bf combine(v8bf lo, v8bf hi) {
    return __builtin_shufflevector(lo, hi, 0,1,2,3,4,5,6,7,8,9,10,11,12,13,14,15);
}

__device__ __forceinline__ v8f wmma_bf16(v16bf a, v16bf b, v8f c) {
    // D = A(16x32 bf16) * B(32x16 bf16) + C(16x16 f32)
    return __builtin_amdgcn_wmma_f32_16x16x32_bf16(
        false, a, false, b, (short)0, c, false, false);
}

// CDNA5 async DMA: global -> LDS, 16B per lane, tracked by ASYNCcnt.
// lds_off is the wave-relative LDS byte address (low 32 bits of the generic
// shared pointer, per the LDS aperture rule addr[31:0]).
__device__ __forceinline__ void async_load_b128(unsigned int lds_off,
                                                const void* gaddr) {
    asm volatile("global_load_async_to_lds_b128 %0, %1, off"
                 :: "v"(lds_off), "v"(gaddr) : "memory");
}

__device__ __forceinline__ void wait_async0() {
    asm volatile("s_wait_asynccnt 0" ::: "memory");
}

// ---------------------------------------------------------------------------
// Kernel 1: qkv = x @ w_qkv  (fp32 in, bf16 out, head-split [B,H,N,64])
// grid (24, 64): x = n-block (3072/128), y = m-block (8192/128); 256 threads
// ---------------------------------------------------------------------------
__global__ void __launch_bounds__(256)
qkv_gemm_kernel(const float* __restrict__ x, const float* __restrict__ w,
                __bf16* __restrict__ qb, __bf16* __restrict__ kb,
                __bf16* __restrict__ vb)
{
    __shared__ unsigned short ldsA[128 * 40];   // [m][k], padded stride 40
    __shared__ unsigned short ldsB[128 * 40];   // [n][k], padded stride 40

    const int tid  = threadIdx.x;
    const int bn   = blockIdx.x * 128;          // 0..3071
    const int bm   = blockIdx.y * 128;          // 0..8191
    const int wid  = tid >> 5, lane = tid & 31;
    const int hf   = lane >> 4, mn = lane & 15;
    const int wrow = wid >> 2,  wcol = wid & 3; // 2x4 wave grid -> 64x32 / wave

    v8f zero = {0.f,0.f,0.f,0.f,0.f,0.f,0.f,0.f};
    v8f acc[4][2];
    #pragma unroll
    for (int i = 0; i < 4; ++i)
        #pragma unroll
        for (int j = 0; j < 2; ++j) acc[i][j] = zero;

    for (int k0 = 0; k0 < DIM; k0 += 32) {
        if (k0 + 32 < DIM) {
            __builtin_prefetch(x + (size_t)(bm + (tid >> 1)) * DIM + k0 + 32, 0, 1);
            __builtin_prefetch(w + (size_t)(k0 + 32 + (tid >> 3)) * 3072 + bn, 0, 1);
        }
        // stage A tile: 128 x 32 fp32 -> bf16 (needs VGPR pass for convert)
        #pragma unroll
        for (int it = 0; it < 4; ++it) {
            int c = tid + it * 256;             // 0..1023 chunks of float4
            int m = c >> 3, k4 = (c & 7) << 2;
            float4 f = *(const float4*)(x + (size_t)(bm + m) * DIM + k0 + k4);
            unsigned int w0 = f2bfu(f.x) | ((unsigned int)f2bfu(f.y) << 16);
            unsigned int w1 = f2bfu(f.z) | ((unsigned int)f2bfu(f.w) << 16);
            *(unsigned int*)&ldsA[m * 40 + k4]     = w0;
            *(unsigned int*)&ldsA[m * 40 + k4 + 2] = w1;
        }
        // stage B tile: 32 x 128 fp32 -> bf16, transposed to [n][k]
        #pragma unroll
        for (int it = 0; it < 4; ++it) {
            int c = tid + it * 256;
            int k = c >> 5, n4 = (c & 31) << 2;
            float4 f = *(const float4*)(w + (size_t)(k0 + k) * 3072 + bn + n4);
            ldsB[(n4 + 0) * 40 + k] = f2bfu(f.x);
            ldsB[(n4 + 1) * 40 + k] = f2bfu(f.y);
            ldsB[(n4 + 2) * 40 + k] = f2bfu(f.z);
            ldsB[(n4 + 3) * 40 + k] = f2bfu(f.w);
        }
        __syncthreads();

        v16bf afr[4], bfr[2];
        #pragma unroll
        for (int i = 0; i < 4; ++i) {
            const unsigned short* base = &ldsA[(wrow * 64 + i * 16 + mn) * 40];
            afr[i] = combine(*(const v8bf*)(base + 8 * hf),
                             *(const v8bf*)(base + 16 + 8 * hf));
        }
        #pragma unroll
        for (int j = 0; j < 2; ++j) {
            const unsigned short* base = &ldsB[(wcol * 32 + j * 16 + mn) * 40];
            bfr[j] = combine(*(const v8bf*)(base + 8 * hf),
                             *(const v8bf*)(base + 16 + 8 * hf));
        }
        #pragma unroll
        for (int i = 0; i < 4; ++i)
            #pragma unroll
            for (int j = 0; j < 2; ++j)
                acc[i][j] = wmma_bf16(afr[i], bfr[j], acc[i][j]);
        __syncthreads();
    }

    // epilogue: scatter into q/k/v [B, H, N, 64] bf16
    #pragma unroll
    for (int i = 0; i < 4; ++i) {
        int gm0 = bm + wrow * 64 + i * 16 + 8 * hf;
        #pragma unroll
        for (int j = 0; j < 2; ++j) {
            int gn    = bn + wcol * 32 + j * 16 + mn;
            int part  = gn >> 10;
            int inner = gn & 1023;
            int head  = inner >> 6, dh = inner & 63;
            __bf16* dst = (part == 0) ? qb : ((part == 1) ? kb : vb);
            unsigned short* du = (unsigned short*)dst;
            #pragma unroll
            for (int r = 0; r < 8; ++r) {
                int row = gm0 + r;
                int b = row >> 11, seq = row & 2047;
                size_t idx = (((size_t)(b * HEADS + head)) * SEQ + seq) * DHEAD + dh;
                du[idx] = f2bfu(acc[i][j][r]);
            }
        }
    }
}

// ---------------------------------------------------------------------------
// Kernel 2: flash attention (online softmax), bf16 WMMA, fp32 accumulate
// grid (16, 64): x = q-block (2048/128), y = b*H + h; 256 threads (8 waves)
// each wave owns 16 query rows; block shares 32-key K/V tiles via LDS.
// K tile is DMA'd straight to LDS with global_load_async_to_lds_b128.
// ---------------------------------------------------------------------------
__global__ void __launch_bounds__(256)
flash_attn_kernel(const __bf16* __restrict__ qb, const __bf16* __restrict__ kb,
                  const __bf16* __restrict__ vb, __bf16* __restrict__ ob)
{
    __shared__ unsigned short ldsK[32 * 40];        // [key][dh]
    __shared__ unsigned short ldsVt[64 * 40];       // [dh][key] (transposed)
    __shared__ unsigned short ldsP[8 * 16 * 40];    // per-wave P restage

    const int tid = threadIdx.x;
    const int wid = tid >> 5, lane = tid & 31;
    const int hf  = lane >> 4, mn = lane & 15;
    const int bh   = blockIdx.y;                    // 0..63
    const int qblk = blockIdx.x;                    // 0..15
    const size_t hb = (size_t)bh * SEQ * DHEAD;
    const int qrow0 = qblk * 128 + wid * 16;

    const unsigned short* qp = (const unsigned short*)qb + hb;
    const unsigned short* kp = (const unsigned short*)kb + hb;
    const unsigned short* vp = (const unsigned short*)vb + hb;

    // Q fragments (A layout), dh chunks [0,32) and [32,64)
    v16bf qf[2];
    #pragma unroll
    for (int c = 0; c < 2; ++c) {
        const unsigned short* p = qp + (size_t)(qrow0 + mn) * DHEAD + 32 * c + 8 * hf;
        qf[c] = combine(*(const v8bf*)p, *(const v8bf*)(p + 16));
    }

    v8f zero = {0.f,0.f,0.f,0.f,0.f,0.f,0.f,0.f};
    v8f o[4];
    float rmax[8], rsum[8];
    #pragma unroll
    for (int t = 0; t < 4; ++t) o[t] = zero;
    #pragma unroll
    for (int r = 0; r < 8; ++r) { rmax[r] = -3.0e38f; rsum[r] = 0.f; }

    unsigned short* myP = &ldsP[wid * 16 * 40];

    const int srow = tid >> 3, sd8 = (tid & 7) << 3;    // staging coords
    const unsigned int kLds =
        (unsigned int)(uintptr_t)&ldsK[srow * 40 + sd8]; // LDS byte offset

    for (int k0 = 0; k0 < SEQ; k0 += 32) {
        __syncthreads();
        // stage K via async DMA (bf16, no transform needed): 16B per lane
        async_load_b128(kLds, kp + (size_t)(k0 + srow) * DHEAD + sd8);
        // stage V transposed (element scatter -> regular path)
        {
            v8bf vv = *(const v8bf*)(vp + (size_t)(k0 + srow) * DHEAD + sd8);
            #pragma unroll
            for (int e = 0; e < 8; ++e)
                ldsVt[(sd8 + e) * 40 + srow] =
                    __builtin_bit_cast(unsigned short, vv[e]);
        }
        wait_async0();
        __syncthreads();

        // S = Q @ K^T  (two 16x16 tiles covering 32 keys)
        v8f s0 = zero, s1 = zero;
        #pragma unroll
        for (int kc = 0; kc < 2; ++kc) {
            const unsigned short* b0 = &ldsK[(mn) * 40 + 32 * kc + 8 * hf];
            const unsigned short* b1 = &ldsK[(16 + mn) * 40 + 32 * kc + 8 * hf];
            v16bf kf0 = combine(*(const v8bf*)b0, *(const v8bf*)(b0 + 16));
            v16bf kf1 = combine(*(const v8bf*)b1, *(const v8bf*)(b1 + 16));
            s0 = wmma_bf16(qf[kc], kf0, s0);
            s1 = wmma_bf16(qf[kc], kf1, s1);
        }

        // online softmax per row; write P (bf16) to private LDS for re-layout
        #pragma unroll
        for (int r = 0; r < 8; ++r) {
            float a0 = s0[r] * 0.125f, a1 = s1[r] * 0.125f;  // 1/sqrt(64)
            float mx = fmaxf(a0, a1);
            mx = fmaxf(mx, __shfl_xor(mx, 8, 16));
            mx = fmaxf(mx, __shfl_xor(mx, 4, 16));
            mx = fmaxf(mx, __shfl_xor(mx, 2, 16));
            mx = fmaxf(mx, __shfl_xor(mx, 1, 16));
            float mnew = fmaxf(rmax[r], mx);
            float fac  = __expf(rmax[r] - mnew);
            rmax[r] = mnew;
            float p0 = __expf(a0 - mnew);
            float p1 = __expf(a1 - mnew);
            float ps = p0 + p1;
            ps += __shfl_xor(ps, 8, 16);
            ps += __shfl_xor(ps, 4, 16);
            ps += __shfl_xor(ps, 2, 16);
            ps += __shfl_xor(ps, 1, 16);
            rsum[r] = rsum[r] * fac + ps;
            #pragma unroll
            for (int t = 0; t < 4; ++t) o[t][r] *= fac;
            int m = 8 * hf + r;
            myP[m * 40 + mn]      = f2bfu(p0);
            myP[m * 40 + 16 + mn] = f2bfu(p1);
        }

        // O += P @ V   (same-wave LDS RAW is in-order; no barrier needed)
        {
            const unsigned short* pa = &myP[mn * 40];
            v16bf pf = combine(*(const v8bf*)(pa + 8 * hf),
                               *(const v8bf*)(pa + 16 + 8 * hf));
            #pragma unroll
            for (int t = 0; t < 4; ++t) {
                const unsigned short* pv = &ldsVt[(t * 16 + mn) * 40 + 8 * hf];
                v16bf vf = combine(*(const v8bf*)pv, *(const v8bf*)(pv + 16));
                o[t] = wmma_bf16(pf, vf, o[t]);
            }
        }
    }

    // epilogue: normalize and store to [B, N, H*64] bf16
    int b = bh >> 4, h = bh & 15;
    unsigned short* op = (unsigned short*)ob;
    #pragma unroll
    for (int t = 0; t < 4; ++t) {
        #pragma unroll
        for (int r = 0; r < 8; ++r) {
            int m = 8 * hf + r;
            float val = o[t][r] / rsum[r];
            size_t idx = ((size_t)(b * SEQ + qrow0 + m)) * DIM + h * DHEAD + t * 16 + mn;
            op[idx] = f2bfu(val);
        }
    }
}

// ---------------------------------------------------------------------------
// Kernel 3: out = ob @ w_out + b_out  (bf16 A via async DMA, fp32 W converted,
// fp32 out). grid (8, 64): x = n-block, y = m-block; 256 threads
// ---------------------------------------------------------------------------
__global__ void __launch_bounds__(256)
out_gemm_kernel(const __bf16* __restrict__ a, const float* __restrict__ w,
                const float* __restrict__ bias, float* __restrict__ out)
{
    __shared__ unsigned short ldsA[128 * 40];
    __shared__ unsigned short ldsB[128 * 40];

    const int tid  = threadIdx.x;
    const int bn   = blockIdx.x * 128;
    const int bm   = blockIdx.y * 128;
    const int wid  = tid >> 5, lane = tid & 31;
    const int hf   = lane >> 4, mn = lane & 15;
    const int wrow = wid >> 2,  wcol = wid & 3;

    v8f zero = {0.f,0.f,0.f,0.f,0.f,0.f,0.f,0.f};
    v8f acc[4][2];
    #pragma unroll
    for (int i = 0; i < 4; ++i)
        #pragma unroll
        for (int j = 0; j < 2; ++j) acc[i][j] = zero;

    const unsigned short* au = (const unsigned short*)a;

    // A staging coords: 128x32 bf16 tile = 512 chunks of 8; 2 per thread
    const int am0 = tid >> 2,          ak0 = (tid & 3) << 3;
    const int am1 = (tid + 256) >> 2,  ak1 = ((tid + 256) & 3) << 3;
    const unsigned int aLds0 = (unsigned int)(uintptr_t)&ldsA[am0 * 40 + ak0];
    const unsigned int aLds1 = (unsigned int)(uintptr_t)&ldsA[am1 * 40 + ak1];

    for (int k0 = 0; k0 < DIM; k0 += 32) {
        // stage A tile via async DMA (already bf16)
        async_load_b128(aLds0, au + (size_t)(bm + am0) * DIM + k0 + ak0);
        async_load_b128(aLds1, au + (size_t)(bm + am1) * DIM + k0 + ak1);
        // stage B tile: 32 x 128 fp32 -> bf16, transposed
        #pragma unroll
        for (int it = 0; it < 4; ++it) {
            int c = tid + it * 256;
            int k = c >> 5, n4 = (c & 31) << 2;
            float4 f = *(const float4*)(w + (size_t)(k0 + k) * DIM + bn + n4);
            ldsB[(n4 + 0) * 40 + k] = f2bfu(f.x);
            ldsB[(n4 + 1) * 40 + k] = f2bfu(f.y);
            ldsB[(n4 + 2) * 40 + k] = f2bfu(f.z);
            ldsB[(n4 + 3) * 40 + k] = f2bfu(f.w);
        }
        wait_async0();
        __syncthreads();

        v16bf afr[4], bfr[2];
        #pragma unroll
        for (int i = 0; i < 4; ++i) {
            const unsigned short* base = &ldsA[(wrow * 64 + i * 16 + mn) * 40];
            afr[i] = combine(*(const v8bf*)(base + 8 * hf),
                             *(const v8bf*)(base + 16 + 8 * hf));
        }
        #pragma unroll
        for (int j = 0; j < 2; ++j) {
            const unsigned short* base = &ldsB[(wcol * 32 + j * 16 + mn) * 40];
            bfr[j] = combine(*(const v8bf*)(base + 8 * hf),
                             *(const v8bf*)(base + 16 + 8 * hf));
        }
        #pragma unroll
        for (int i = 0; i < 4; ++i)
            #pragma unroll
            for (int j = 0; j < 2; ++j)
                acc[i][j] = wmma_bf16(afr[i], bfr[j], acc[i][j]);
        __syncthreads();
    }

    #pragma unroll
    for (int j = 0; j < 2; ++j) {
        int gn = bn + wcol * 32 + j * 16 + mn;
        float bb = bias[gn];
        #pragma unroll
        for (int i = 0; i < 4; ++i) {
            int gm0 = bm + wrow * 64 + i * 16 + 8 * hf;
            #pragma unroll
            for (int r = 0; r < 8; ++r)
                out[(size_t)(gm0 + r) * DIM + gn] = acc[i][j][r] + bb;
        }
    }
}

// ---------------------------------------------------------------------------
// Launch
// ---------------------------------------------------------------------------
extern "C" void kernel_launch(void* const* d_in, const int* in_sizes, int n_in,
                              void* d_out, int out_size, void* d_ws, size_t ws_size,
                              hipStream_t stream)
{
    (void)in_sizes; (void)n_in; (void)out_size; (void)ws_size;
    const float* x     = (const float*)d_in[0];
    const float* w_qkv = (const float*)d_in[1];
    const float* w_out = (const float*)d_in[2];
    const float* b_out = (const float*)d_in[3];
    float* out = (float*)d_out;

    // workspace: q, k, v in [B,H,N,64] bf16 + attn-out in [B,N,1024] bf16
    __bf16* ws = (__bf16*)d_ws;
    const size_t seg = (size_t)BATCH * HEADS * SEQ * DHEAD;  // 8M elements
    __bf16* qb = ws;
    __bf16* kb = ws + seg;
    __bf16* vb = ws + 2 * seg;
    __bf16* ob = ws + 3 * seg;

    qkv_gemm_kernel<<<dim3(24, 64), 256, 0, stream>>>(x, w_qkv, qb, kb, vb);
    flash_attn_kernel<<<dim3(16, 64), 256, 0, stream>>>(qb, kb, vb, ob);
    out_gemm_kernel<<<dim3(8, 64), 256, 0, stream>>>(ob, w_out, b_out, out);
}
